// MultiHeadAttention_16836271800572
// MI455X (gfx1250) — matrix operand order repaired and checked by reading, not compiled, not run on hardware
//
#include <hip/hip_runtime.h>

// ---------------------------------------------------------------------------
// MultiHeadAttention for MI455X (gfx1250, wave32, WMMA bf16 -> f32 accum)
// B=4, N=2048, D=1024, H=16, DH=64
// ---------------------------------------------------------------------------

#define BATCH 4
#define NSEQ  2048
#define DMODEL 1024
#define NHEAD 16
#define DHEAD 64

typedef __attribute__((ext_vector_type(16))) __bf16 v16bf;
typedef __attribute__((ext_vector_type(8)))  __bf16 v8bf;
typedef __attribute__((ext_vector_type(8)))  float  v8f;

__device__ __forceinline__ __bf16 f2bf(float f) {
    unsigned u = __builtin_bit_cast(unsigned, f);
    unsigned r = u + 0x7FFFu + ((u >> 16) & 1u);   // round-to-nearest-even
    unsigned short h = (unsigned short)(r >> 16);
    return __builtin_bit_cast(__bf16, h);
}

// Combine two contiguous 8-element chunks into a 16-element fragment register.
__device__ __forceinline__ v16bf ld_frag(const __bf16* p0, const __bf16* p1) {
    v8bf a = *(const v8bf*)p0;
    v8bf b = *(const v8bf*)p1;
    v16bf r;
#pragma unroll
    for (int i = 0; i < 8; ++i) { r[i] = a[i]; r[8 + i] = b[i]; }
    return r;
}

__device__ __forceinline__ v8f wmma_bf16(v16bf a, v16bf b, v8f c) {
    return __builtin_amdgcn_wmma_f32_16x16x32_bf16(
        /*neg_a=*/false, a, /*neg_b=*/false, b,
        /*c_mod=*/(short)0, c, /*reuse_a=*/false, /*reuse_b=*/false);
}

__device__ __forceinline__ float redmax16(float v) {
    v = fmaxf(v, __shfl_xor(v, 1, 32));
    v = fmaxf(v, __shfl_xor(v, 2, 32));
    v = fmaxf(v, __shfl_xor(v, 4, 32));
    v = fmaxf(v, __shfl_xor(v, 8, 32));
    return v;
}
__device__ __forceinline__ float redsum16(float v) {
    v += __shfl_xor(v, 1, 32);
    v += __shfl_xor(v, 2, 32);
    v += __shfl_xor(v, 4, 32);
    v += __shfl_xor(v, 8, 32);
    return v;
}

// ---------------------------------------------------------------------------
// Projection GEMM: Y[8192x1024] = X[8192x1024] @ W[1024x1024]   (bf16 WMMA)
// mode 0: write bf16 (B,H,N,DH)     (Q-proj, K-proj)
// mode 1: write bf16 (B,H,DH,N)     (V-proj, transposed for PV operand)
// mode 2: write f32  (B,N,D)        (output projection)
// Block: 256 threads (8 waves). Block tile 128x128; wave tile 64x32.
// ---------------------------------------------------------------------------
#define LDSTR 40   // padded LDS row stride in bf16 elements (kills bank conflicts)

__global__ __launch_bounds__(256)
void proj_kernel(const float* __restrict__ X, const float* __restrict__ W,
                 void* __restrict__ out, int mode) {
    __shared__ __bf16 lA[128 * LDSTR];   // X tile, rows = m, cols = k(32)
    __shared__ __bf16 lB[128 * LDSTR];   // W^T tile, rows = n, cols = k(32)

    const int tid  = threadIdx.x;
    const int lane = tid & 31;
    const int wave = tid >> 5;
    const int wm   = wave & 1;        // M half (0..1) -> 64 rows
    const int wn   = wave >> 1;       // N quarter (0..3) -> 32 cols
    const int lh   = lane >> 4;       // lane half
    const int ll   = lane & 15;
    const int m0   = blockIdx.x * 128;
    const int n0   = blockIdx.y * 128;

    v8f acc[4][2] = {};

    for (int kt = 0; kt < DMODEL; kt += 32) {
        __syncthreads();
        // --- stage X tile (128 x 32 f32 -> bf16 LDS) ---
#pragma unroll
        for (int i = 0; i < 4; ++i) {
            int idx = tid + i * 256;          // 1024 float4 total
            int row = idx >> 3;
            int c4  = (idx & 7) * 4;
            float4 v = *(const float4*)(X + (size_t)(m0 + row) * DMODEL + kt + c4);
            __bf16* d = &lA[row * LDSTR + c4];
            d[0] = f2bf(v.x); d[1] = f2bf(v.y); d[2] = f2bf(v.z); d[3] = f2bf(v.w);
        }
        // --- stage W^T tile (32 x 128 f32 -> transposed bf16 LDS) ---
#pragma unroll
        for (int i = 0; i < 4; ++i) {
            int idx = tid + i * 256;
            int kk  = idx >> 5;               // 0..31
            int c4  = (idx & 31) * 4;         // 0..124
            float4 v = *(const float4*)(W + (size_t)(kt + kk) * DMODEL + n0 + c4);
            lB[(c4 + 0) * LDSTR + kk] = f2bf(v.x);
            lB[(c4 + 1) * LDSTR + kk] = f2bf(v.y);
            lB[(c4 + 2) * LDSTR + kk] = f2bf(v.z);
            lB[(c4 + 3) * LDSTR + kk] = f2bf(v.w);
        }
        __syncthreads();

        // --- fragments + 8 WMMA ---
        v16bf af[4], bfr[2];
#pragma unroll
        for (int ms = 0; ms < 4; ++ms) {
            // A 16x32: lane half 0 -> K 0-7 & 16-23 ; half 1 -> K 8-15 & 24-31
            const __bf16* p = &lA[(wm * 64 + ms * 16 + ll) * LDSTR + lh * 8];
            af[ms] = ld_frag(p, p + 16);
        }
#pragma unroll
        for (int ns = 0; ns < 2; ++ns) {
            // B 32x16: lane = column, half 0 -> K 0-7 & 16-23 ; half 1 -> K 8-15 & 24-31
            const __bf16* p = &lB[(wn * 32 + ns * 16 + ll) * LDSTR + lh * 8];
            bfr[ns] = ld_frag(p, p + 16);
        }
#pragma unroll
        for (int ms = 0; ms < 4; ++ms)
#pragma unroll
            for (int ns = 0; ns < 2; ++ns)
                acc[ms][ns] = wmma_bf16(af[ms], bfr[ns], acc[ms][ns]);
    }

    // --- epilogue: C layout row = r + 8*lh, col = ll ---
#pragma unroll
    for (int ms = 0; ms < 4; ++ms) {
#pragma unroll
        for (int ns = 0; ns < 2; ++ns) {
#pragma unroll
            for (int r = 0; r < 8; ++r) {
                int m = m0 + wm * 64 + ms * 16 + r + 8 * lh;
                int c = n0 + wn * 32 + ns * 16 + ll;
                float v = acc[ms][ns][r];
                if (mode == 2) {
                    ((float*)out)[(size_t)m * DMODEL + c] = v;
                } else {
                    int b = m >> 11, n = m & (NSEQ - 1);
                    int h = c >> 6, dh = c & 63;
                    size_t o = (mode == 0)
                        ? ((((size_t)b * NHEAD + h) * NSEQ + n) * DHEAD + dh)
                        : ((((size_t)b * NHEAD + h) * DHEAD + dh) * NSEQ + n);
                    ((__bf16*)out)[o] = f2bf(v);
                }
            }
        }
    }
}

// ---------------------------------------------------------------------------
// Flash attention: one wave = 16 queries; 32 keys per iteration.
// Qh,Kh bf16 (B,H,N,64) ; Vt bf16 (B,H,64,N) ; Oc f32 (B,N,D).
// Block = 128 threads (4 waves) covering 64 queries of one (b,h).
// ---------------------------------------------------------------------------
__global__ __launch_bounds__(128)
void attn_kernel(const void* __restrict__ Qh_, const void* __restrict__ Kh_,
                 const void* __restrict__ Vt_, const int* __restrict__ valid_lens,
                 float* __restrict__ Oc) {
    const __bf16* Qh = (const __bf16*)Qh_;
    const __bf16* Kh = (const __bf16*)Kh_;
    const __bf16* Vt = (const __bf16*)Vt_;

    __shared__ __bf16 lP[4][16 * LDSTR];   // per-wave P tile (16 x 32)

    const int wave = threadIdx.x >> 5;
    const int lane = threadIdx.x & 31;
    const int lh   = lane >> 4;
    const int ll   = lane & 15;

    const int qt = blockIdx.x & (NSEQ / 64 - 1);   // 32 q-tiles of 64
    const int bh = blockIdx.x / (NSEQ / 64);       // 0..63
    const int b  = bh >> 4;
    const int h  = bh & 15;
    const int q0 = qt * 64 + wave * 16;

    const __bf16* Qp = Qh + ((size_t)bh * NSEQ + q0) * DHEAD;
    const __bf16* Kp = Kh + (size_t)bh * NSEQ * DHEAD;
    const __bf16* Vp = Vt + (size_t)bh * DHEAD * NSEQ;
    const int vl = valid_lens[b];

    // Q fragments (A layout), reused all iterations
    v16bf qf[2];
#pragma unroll
    for (int s = 0; s < 2; ++s) {
        const __bf16* p = Qp + ll * DHEAD + s * 32 + lh * 8;
        qf[s] = ld_frag(p, p + 16);
    }

    v8f acc[4] = {};
    float mr[8], lr[8];
#pragma unroll
    for (int r = 0; r < 8; ++r) { mr[r] = -1e30f; lr[r] = 0.f; }

    __bf16* P = lP[wave];
    const int nk = ((vl + 31) >> 5) << 5;   // keys rounded up to 32; rest are exact zeros

    for (int kb = 0; kb < nk; kb += 32) {
        // ---- S = Q K^T for 2 key sub-tiles (4 WMMA) ----
        v8f S0 = {}, S1 = {};
#pragma unroll
        for (int s = 0; s < 2; ++s) {
            const __bf16* k0p = Kp + (size_t)(kb + ll) * DHEAD + s * 32 + lh * 8;
            const __bf16* k1p = Kp + (size_t)(kb + 16 + ll) * DHEAD + s * 32 + lh * 8;
            S0 = wmma_bf16(qf[s], ld_frag(k0p, k0p + 16), S0);
            S1 = wmma_bf16(qf[s], ld_frag(k1p, k1p + 16), S1);
        }

        // ---- scale, mask, online softmax stats ----
        const bool v0 = (kb + ll) < vl;
        const bool v1 = (kb + 16 + ll) < vl;
        float sc[8];
#pragma unroll
        for (int r = 0; r < 8; ++r) {
            float s0 = v0 ? S0[r] * 0.125f : -1e20f;   // 1/sqrt(64)
            float s1 = v1 ? S1[r] * 0.125f : -1e20f;
            float mx = redmax16(fmaxf(s0, s1));
            float mn = fmaxf(mr[r], mx);
            float scale = __expf(mr[r] - mn);
            float p0 = __expf(s0 - mn);
            float p1 = __expf(s1 - mn);
            lr[r] = lr[r] * scale + redsum16(p0 + p1);
            mr[r] = mn;
            sc[r] = scale;
            // C-layout -> LDS (row-major 16x32 P tile)
            P[(r + 8 * lh) * LDSTR + ll]      = f2bf(p0);
            P[(r + 8 * lh) * LDSTR + 16 + ll] = f2bf(p1);
        }
#pragma unroll
        for (int d = 0; d < 4; ++d)
#pragma unroll
            for (int r = 0; r < 8; ++r) acc[d][r] *= sc[r];

        // ---- O += P V  (A frag from LDS, 4 WMMA) ----
        const __bf16* pp = P + ll * LDSTR + lh * 8;
        v16bf pf = ld_frag(pp, pp + 16);
#pragma unroll
        for (int d = 0; d < 4; ++d) {
            const __bf16* vp = Vp + (size_t)(d * 16 + ll) * NSEQ + kb + lh * 8;
            acc[d] = wmma_bf16(pf, ld_frag(vp, vp + 16), acc[d]);
        }
    }

    // ---- normalize and write concat layout (B,N,D) f32 ----
#pragma unroll
    for (int r = 0; r < 8; ++r) {
        float inv = 1.f / lr[r];
        int row = q0 + r + 8 * lh;
#pragma unroll
        for (int d = 0; d < 4; ++d) {
            int c = h * DHEAD + d * 16 + ll;
            Oc[((size_t)b * NSEQ + row) * DMODEL + c] = acc[d][r] * inv;
        }
    }
}

// ---------------------------------------------------------------------------
extern "C" void kernel_launch(void* const* d_in, const int* in_sizes, int n_in,
                              void* d_out, int out_size, void* d_ws, size_t ws_size,
                              hipStream_t stream) {
    const float* Qb = (const float*)d_in[0];
    const float* Kb = (const float*)d_in[1];
    const float* Vb = (const float*)d_in[2];
    const int*   vl = (const int*)d_in[3];
    const float* Wq = (const float*)d_in[4];
    const float* Wk = (const float*)d_in[5];
    const float* Wv = (const float*)d_in[6];
    const float* Wo = (const float*)d_in[7];

    char* ws = (char*)d_ws;
    // bf16 Qh/Kh: (B,H,N,64) = 8M elem = 16MB each ; Vt: (B,H,64,N) 16MB ; Oc f32 32MB
    void*  Qh = (void*)(ws);
    void*  Kh = (void*)(ws + (size_t)16 * 1024 * 1024);
    void*  Vt = (void*)(ws + (size_t)32 * 1024 * 1024);
    float* Oc = (float*)(ws + (size_t)48 * 1024 * 1024);

    dim3 pg(64, 8), pb(256);
    proj_kernel<<<pg, pb, 0, stream>>>(Qb, Wq, Qh, 0);
    proj_kernel<<<pg, pb, 0, stream>>>(Kb, Wk, Kh, 0);
    proj_kernel<<<pg, pb, 0, stream>>>(Vb, Wv, Vt, 1);

    attn_kernel<<<dim3(BATCH * NHEAD * (NSEQ / 64)), dim3(128), 0, stream>>>(
        Qh, Kh, Vt, vl, Oc);

    proj_kernel<<<pg, pb, 0, stream>>>(Oc, Wo, d_out, 2);
}
